// W8A16LL_1563368096495
// MI455X (gfx1250) — matrix operand order repaired and checked by reading, not compiled, not run on hardware
//
#include <hip/hip_runtime.h>
#include <stdint.h>

typedef _Float16 v16h  __attribute__((ext_vector_type(16)));
typedef _Float16 half8 __attribute__((ext_vector_type(8)));
typedef float    v8f   __attribute__((ext_vector_type(8)));
typedef float    f32x4 __attribute__((ext_vector_type(4)));
typedef int8_t   i8x16 __attribute__((ext_vector_type(16)));

#define M_TOT 4096   // B*S = 2*2048
#define N_TOT 4096   // OUT_F
#define K_TOT 4096   // IN_F

#define BM 128
#define BN 128
#define BK 64
#define LDS_STRIDE (BK + 8)   // 72 halves = 144 B: keeps 16B alignment, staggers banks

__global__ __launch_bounds__(256)
void w8a16_gemm_wmma_f16(const float* __restrict__ X,
                         const int8_t* __restrict__ Wq,
                         const float* __restrict__ scales,
                         const float* __restrict__ bias,
                         float* __restrict__ Out)
{
    __shared__ __attribute__((aligned(16))) _Float16 Alds[BM][LDS_STRIDE];
    __shared__ __attribute__((aligned(16))) _Float16 Blds[BN][LDS_STRIDE];

    const int tid  = threadIdx.x;
    const int lane = tid & 31;
    const int wave = tid >> 5;
    const int wm   = wave & 3;      // 4 waves along M -> 32 rows each
    const int wn   = wave >> 2;     // 2 waves along N -> 64 cols each
    const int hs   = lane >> 4;     // half-wave select (WMMA operand layout)
    const int l15  = lane & 15;

    const int block_m = blockIdx.y * BM;
    const int block_n = blockIdx.x * BN;

    // Staging map: each thread owns 32 contiguous K-elements of one row
    const int srow = tid >> 1;
    const int scol = (tid & 1) * 32;
    const float*  gA = X  + (size_t)(block_m + srow) * K_TOT + scol;
    const int8_t* gW = Wq + (size_t)(block_n + srow) * K_TOT + scol;

    v8f acc[2][4];
#pragma unroll
    for (int tm = 0; tm < 2; ++tm)
#pragma unroll
        for (int tn = 0; tn < 4; ++tn)
            acc[tm][tn] = (v8f){};

    for (int k0 = 0; k0 < K_TOT; k0 += BK) {
        // ---- stage A tile: fp32 -> f16 into LDS (128 B contiguous per thread) ----
#pragma unroll
        for (int c = 0; c < 4; ++c) {
            f32x4 f0 = ((const f32x4*)gA)[2 * c + 0];
            f32x4 f1 = ((const f32x4*)gA)[2 * c + 1];
            half8 h;
            h[0] = (_Float16)f0[0]; h[1] = (_Float16)f0[1];
            h[2] = (_Float16)f0[2]; h[3] = (_Float16)f0[3];
            h[4] = (_Float16)f1[0]; h[5] = (_Float16)f1[1];
            h[6] = (_Float16)f1[2]; h[7] = (_Float16)f1[3];
            *(half8*)&Alds[srow][scol + 8 * c] = h;
        }
        // ---- stage B tile: int8 -> f16 into LDS (exact dequant) ----
#pragma unroll
        for (int c = 0; c < 2; ++c) {
            i8x16 wv = ((const i8x16*)gW)[c];
            half8 h0, h1;
#pragma unroll
            for (int i = 0; i < 8; ++i) {
                h0[i] = (_Float16)(short)wv[i];
                h1[i] = (_Float16)(short)wv[8 + i];
            }
            *(half8*)&Blds[srow][scol + 16 * c]     = h0;
            *(half8*)&Blds[srow][scol + 16 * c + 8] = h1;
        }
        gA += BK;
        gW += BK;
        if (k0 + BK < K_TOT) {          // pull next K-tile toward L2/L0
            __builtin_prefetch(gA, 0, 3);
            __builtin_prefetch(gW, 0, 3);
        }
        __syncthreads();

        // ---- compute: 2 k-steps x (2x4) 16x16 tiles = 16 WMMAs per stage ----
#pragma unroll
        for (int kk = 0; kk < BK; kk += 32) {
            v16h afrag[2], bfrag[4];
#pragma unroll
            for (int tm = 0; tm < 2; ++tm) {
                // A 16x32 f16 layout: lanes 0-15 row M=l15 hold K=[kk..kk+7],[kk+16..kk+23]
                //                     lanes 16-31 hold K runs shifted by 8
                const _Float16* pa = &Alds[wm * 32 + tm * 16 + l15][kk + 8 * hs];
                half8 lo = *(const half8*)pa;
                half8 hi = *(const half8*)(pa + 16);
                afrag[tm] = __builtin_shufflevector(lo, hi,
                    0, 1, 2, 3, 4, 5, 6, 7, 8, 9, 10, 11, 12, 13, 14, 15);
            }
#pragma unroll
            for (int tn = 0; tn < 4; ++tn) {
                // B 32x16 f16 layout: column N=l15; lanes 0-15 hold K=[kk..kk+7],[kk+16..kk+23]
                //                     lanes 16-31 hold K runs shifted by 8
                const _Float16* pb = &Blds[wn * 64 + tn * 16 + l15][kk + 8 * hs];
                half8 lo = *(const half8*)pb;
                half8 hi = *(const half8*)(pb + 16);
                bfrag[tn] = __builtin_shufflevector(lo, hi,
                    0, 1, 2, 3, 4, 5, 6, 7, 8, 9, 10, 11, 12, 13, 14, 15);
            }
#pragma unroll
            for (int tm = 0; tm < 2; ++tm)
#pragma unroll
                for (int tn = 0; tn < 4; ++tn)
                    acc[tm][tn] = __builtin_amdgcn_wmma_f32_16x16x32_f16(
                        false, afrag[tm], false, bfrag[tn],
                        (short)0, acc[tm][tn], false, false);
        }
        __syncthreads();
    }

    // ---- epilogue: per-output-channel scale + bias, fp32 store ----
#pragma unroll
    for (int tm = 0; tm < 2; ++tm) {
#pragma unroll
        for (int tn = 0; tn < 4; ++tn) {
            const int row0 = block_m + wm * 32 + tm * 16 + hs * 8; // C/D: M = vgpr + 8*hs
            const int col  = block_n + wn * 64 + tn * 16 + l15;    // C/D: N = l15
            const float s  = scales[col];
            const float bb = bias[col];
            v8f c = acc[tm][tn];
#pragma unroll
            for (int r = 0; r < 8; ++r)
                Out[(size_t)(row0 + r) * N_TOT + col] = c[r] * s + bb;
        }
    }
}

extern "C" void kernel_launch(void* const* d_in, const int* in_sizes, int n_in,
                              void* d_out, int out_size, void* d_ws, size_t ws_size,
                              hipStream_t stream) {
    (void)in_sizes; (void)n_in; (void)out_size; (void)d_ws; (void)ws_size;
    const float*  X      = (const float*)d_in[0];
    const int8_t* Wq     = (const int8_t*)d_in[1];
    const float*  scales = (const float*)d_in[2];
    const float*  bias   = (const float*)d_in[3];
    float*        Out    = (float*)d_out;

    dim3 grid(N_TOT / BN, M_TOT / BM);   // 32 x 32 workgroups
    dim3 block(256);                     // 8 wave32 per WG
    w8a16_gemm_wmma_f16<<<grid, block, 0, stream>>>(X, Wq, scales, bias, Out);
}
